// SuperPoint_70437463654407
// MI455X (gfx1250) — compile-verified, matrix operand order checked by reading, not run on hardware
//
#include <hip/hip_runtime.h>
#include <hip/hip_bf16.h>
#include <math.h>

typedef __attribute__((ext_vector_type(16))) _Float16 v16h;
typedef __attribute__((ext_vector_type(8)))  _Float16 h8;
typedef __attribute__((ext_vector_type(8)))  float    v8f;

#define NBINS 4096
#define KTOP  1024

static __device__ __forceinline__ int iclamp(int v, int lo, int hi) {
  return v < lo ? lo : (v > hi ? hi : v);
}

// ---------------------------------------------------------------------------
// conv1a: Cin=1, direct VALU conv, 3x3 pad 1, ReLU, output NHWC f16 (C=64)
// ---------------------------------------------------------------------------
__global__ void k_conv1(const float* __restrict__ img, const float* __restrict__ w,
                        const float* __restrict__ bias, _Float16* __restrict__ out,
                        int H, int W) {
  int idx = blockIdx.x * blockDim.x + threadIdx.x;
  int npix = H * W;
  int b = idx / npix;
  if (b >= 2) return;
  int p = idx - b * npix;
  int y = p / W, x = p - y * W;
  const float* im = img + (size_t)b * npix;
  float patch[9];
#pragma unroll
  for (int t = 0; t < 9; ++t) {
    int yy = y + t / 3 - 1, xx = x + t % 3 - 1;
    patch[t] = (yy >= 0 && yy < H && xx >= 0 && xx < W) ? im[(size_t)yy * W + xx] : 0.f;
  }
  _Float16* o = out + ((size_t)b * npix + p) * 64;
  for (int co = 0; co < 64; ++co) {
    float acc = bias[co];
#pragma unroll
    for (int t = 0; t < 9; ++t) acc = fmaf(w[co * 9 + t], patch[t], acc);
    o[co] = (_Float16)fmaxf(acc, 0.f);
  }
}

// ---------------------------------------------------------------------------
// Pack OIHW f32 weights into WMMA B-fragments (f16).
// frag element ((kpos*nChunks + c)*coTP + coT)*32 + lane  [16 halves each]
// lane holds column n=lane&15; K slots per CDNA5 16-bit B layout:
//   lanes 0-15: K = c*32 + {0..7, 16..23};  lanes 16-31: K = c*32 + {8..15, 24..31}
// coTP is the cout-tile count padded to a multiple of 4; pad tiles are zero.
// ---------------------------------------------------------------------------
__global__ void k_prep_wfrag(const float* __restrict__ w, _Float16* __restrict__ frag,
                             int Cout, int Cin, int ks, int coTP) {
  int nChunks = Cin / 32;
  int kpos = blockIdx.x / nChunks;
  int c    = blockIdx.x - kpos * nChunks;
  int ky = kpos / ks, kx = kpos - ky * ks;
  int coT = blockIdx.y;
  int lane = threadIdx.x;
  int n = lane & 15, kh = lane >> 4;
  int co = coT * 16 + n;
  _Float16* dst = frag + ((((size_t)kpos * nChunks + c) * coTP + coT) * 32 + lane) * 16;
  for (int e = 0; e < 16; ++e) {
    int kl = kh * 8 + ((e < 8) ? e : (e + 8));  // e<8 -> kh*8+e ; e>=8 -> kh*8+16+(e-8)
    int ci = c * 32 + kl;
    float v = 0.f;
    if (co < Cout && ci < Cin)
      v = w[(((size_t)co * Cin + ci) * ks + ky) * ks + kx];
    dst[e] = (_Float16)v;
  }
}

// ---------------------------------------------------------------------------
// WMMA implicit-GEMM conv. NHWC f16 in/out.
// Block = 128 threads (4 waves); each wave computes 32 pixels x 64 couts
// (8 accumulator tiles: 2 pixel halves x 4 cout tiles).
// B fragments for the block's 4 cout tiles are staged per-kpos into LDS via
// GLOBAL_LOAD_ASYNC_TO_LDS_B128 (ASYNCcnt) and shared by all 4 waves; A
// fragments come from global as two aligned 16B loads per lane per 32-K chunk
// (contiguous NHWC channels).
// ---------------------------------------------------------------------------
__global__ void k_conv_wmma(const _Float16* __restrict__ in, const _Float16* __restrict__ frag,
                            const float* __restrict__ bias, _Float16* __restrict__ out,
                            int H, int W, int Cin, int Cout, int ks, int pad, int relu,
                            int coTP) {
  __shared__ _Float16 smem[8 * 4 * 512];  // up to 8 chunks x 4 cout tiles x 512 halves (32KB)
  int tid  = threadIdx.x;
  int lane = tid & 31;
  int wave = tid >> 5;
  int tilesX = W >> 5;                    // 32-pixel tiles per row
  int tile = blockIdx.x * 4 + wave;
  int y  = tile / tilesX;
  int x0 = (tile - y * tilesX) << 5;
  int coT0 = blockIdx.y << 2;
  int b = blockIdx.z;
  int row = lane & 15, kh = lane >> 4;
  int nChunks = Cin >> 5;
  int kk = ks * ks;
  v8f acc0a = {}, acc1a = {}, acc2a = {}, acc3a = {};
  v8f acc0b = {}, acc1b = {}, acc2b = {}, acc3b = {};
  const _Float16* base = in + (size_t)b * H * W * Cin;
  // LDS byte address of smem derived from the pointer itself: this both feeds
  // the async-load VDST operand and makes smem escape into the asm, so the
  // compiler must keep the ds reads of the staged fragments live.
  unsigned ldsBase = (unsigned)(size_t)(void*)smem;
  for (int kpos = 0; kpos < kk; ++kpos) {
    // ---- async stage of this kpos's B fragments into LDS (ASYNCcnt) ----
    __syncthreads();   // protect previous kpos's LDS reads before overwrite
    for (int c = 0; c < nChunks; ++c) {
      const char* src = (const char*)(frag + (((size_t)kpos * nChunks + c) * coTP + coT0) * 512);
      unsigned ldsOff = ldsBase + (unsigned)(c * 4096) + (unsigned)tid * 16;
      const char* g0 = src + tid * 16;
      asm volatile("global_load_async_to_lds_b128 %0, %1, off"
                   :: "v"(ldsOff), "v"(g0) : "memory");
      unsigned ldsOff2 = ldsOff + 2048;
      const char* g1 = g0 + 2048;
      asm volatile("global_load_async_to_lds_b128 %0, %1, off"
                   :: "v"(ldsOff2), "v"(g1) : "memory");
    }
    asm volatile("s_wait_asynccnt 0" ::: "memory");  // our wave's async writes done
    __syncthreads();                                 // all waves' writes visible

    int yy  = y + kpos / ks - pad;
    int xx0 = x0 + row + kpos % ks - pad;
    int xx1 = xx0 + 16;
    bool iny = (yy >= 0) && (yy < H);
    bool in0 = iny && (xx0 >= 0) && (xx0 < W);
    bool in1 = iny && (xx1 >= 0) && (xx1 < W);
    const _Float16* s0 = base + ((size_t)yy * W + xx0) * Cin;
    const _Float16* s1 = base + ((size_t)yy * W + xx1) * Cin;
    for (int c = 0; c < nChunks; ++c) {
      int kb = (c << 5) + kh * 8;
      v16h a0 = {}, a1 = {};
      if (in0) {
        h8 lo = *(const h8*)(s0 + kb);
        h8 hi = *(const h8*)(s0 + kb + 16);
        __builtin_prefetch(s0 + kb + 32, 0, 1);  // global_prefetch_b8
#pragma unroll
        for (int e = 0; e < 8; ++e) { a0[e] = lo[e]; a0[8 + e] = hi[e]; }
      }
      if (in1) {
        h8 lo = *(const h8*)(s1 + kb);
        h8 hi = *(const h8*)(s1 + kb + 16);
#pragma unroll
        for (int e = 0; e < 8; ++e) { a1[e] = lo[e]; a1[8 + e] = hi[e]; }
      }
      const v16h* fb = (const v16h*)(smem + c * 2048) + lane;
      v16h b0 = fb[0];
      v16h b1 = fb[32];
      v16h b2 = fb[64];
      v16h b3 = fb[96];
      acc0a = __builtin_amdgcn_wmma_f32_16x16x32_f16(false, a0, false, b0, (short)0, acc0a, false, false);
      acc0b = __builtin_amdgcn_wmma_f32_16x16x32_f16(false, a1, false, b0, (short)0, acc0b, false, false);
      acc1a = __builtin_amdgcn_wmma_f32_16x16x32_f16(false, a0, false, b1, (short)0, acc1a, false, false);
      acc1b = __builtin_amdgcn_wmma_f32_16x16x32_f16(false, a1, false, b1, (short)0, acc1b, false, false);
      acc2a = __builtin_amdgcn_wmma_f32_16x16x32_f16(false, a0, false, b2, (short)0, acc2a, false, false);
      acc2b = __builtin_amdgcn_wmma_f32_16x16x32_f16(false, a1, false, b2, (short)0, acc2b, false, false);
      acc3a = __builtin_amdgcn_wmma_f32_16x16x32_f16(false, a0, false, b3, (short)0, acc3a, false, false);
      acc3b = __builtin_amdgcn_wmma_f32_16x16x32_f16(false, a1, false, b3, (short)0, acc3b, false, false);
    }
  }
  // D layout: VGPR r -> M = r + kh*8 ; N = lane&15
  int n = lane & 15;
  size_t rowbase = ((size_t)b * H + y) * W;
  auto store_tile = [&](const v8f& acc, int coT, int xb) {
    int co = coT * 16 + n;
    if (co >= Cout) return;
    float bv = bias[co];
#pragma unroll
    for (int r = 0; r < 8; ++r) {
      float v = acc[r] + bv;
      if (relu) v = fmaxf(v, 0.f);
      out[(rowbase + (xb + r + kh * 8)) * Cout + co] = (_Float16)v;
    }
  };
  store_tile(acc0a, coT0,     x0);  store_tile(acc0b, coT0,     x0 + 16);
  store_tile(acc1a, coT0 + 1, x0);  store_tile(acc1b, coT0 + 1, x0 + 16);
  store_tile(acc2a, coT0 + 2, x0);  store_tile(acc2b, coT0 + 2, x0 + 16);
  store_tile(acc3a, coT0 + 3, x0);  store_tile(acc3b, coT0 + 3, x0 + 16);
}

// ---------------------------------------------------------------------------
// 2x2 max pool, NHWC f16
// ---------------------------------------------------------------------------
__global__ void k_pool(const _Float16* __restrict__ in, _Float16* __restrict__ out,
                       int Ho, int Wo, int C) {
  size_t i = (size_t)blockIdx.x * blockDim.x + threadIdx.x;
  size_t tot = (size_t)2 * Ho * Wo * C;
  if (i >= tot) return;
  int c = (int)(i % C);
  size_t r = i / C;
  int x = (int)(r % Wo); r /= Wo;
  int y = (int)(r % Ho);
  int b = (int)(r / Ho);
  int Wi = Wo * 2;
  const _Float16* p = in + (((size_t)b * (Ho * 2) + y * 2) * Wi + x * 2) * C + c;
  size_t rs = (size_t)Wi * C;
  float m = fmaxf(fmaxf((float)p[0], (float)p[C]),
                  fmaxf((float)p[rs], (float)p[rs + C]));
  out[i] = (_Float16)m;
}

// ---------------------------------------------------------------------------
// softmax over 65 ch, drop last, 8x8 pixel shuffle into full-res f32 score map
// ---------------------------------------------------------------------------
__global__ void k_softmax_shuffle(const _Float16* __restrict__ raw, float* __restrict__ S,
                                  int hc, int wc) {
  int i = blockIdx.x * blockDim.x + threadIdx.x;
  int tot = 2 * hc * wc;
  if (i >= tot) return;
  int x = i % wc;
  int y = (i / wc) % hc;
  int b = i / (wc * hc);
  const _Float16* p = raw + (size_t)i * 65;
  float mx = -1e30f;
  for (int c = 0; c < 65; ++c) mx = fmaxf(mx, (float)p[c]);
  float e[65];
  float sum = 0.f;
  for (int c = 0; c < 65; ++c) { e[c] = __expf((float)p[c] - mx); sum += e[c]; }
  float inv = 1.f / sum;
  int Wf = wc * 8;
  for (int c = 0; c < 64; ++c) {
    int yy = y * 8 + (c >> 3), xx = x * 8 + (c & 7);
    S[((size_t)b * (hc * 8) + yy) * Wf + xx] = e[c] * inv;
  }
}

// ---------------------------------------------------------------------------
// separable 9x9 max (r=4), SAME with -inf identity => skip OOB
// ---------------------------------------------------------------------------
__global__ void k_rowmax(const float* __restrict__ in, float* __restrict__ out, int H, int W) {
  size_t i = (size_t)blockIdx.x * blockDim.x + threadIdx.x;
  size_t tot = (size_t)2 * H * W;
  if (i >= tot) return;
  int x = (int)(i % W);
  size_t r = i / W;
  const float* p = in + r * W;
  float m = -INFINITY;
#pragma unroll
  for (int d = -4; d <= 4; ++d) {
    int xx = x + d;
    if (xx >= 0 && xx < W) m = fmaxf(m, p[xx]);
  }
  out[i] = m;
}

__global__ void k_colmax(const float* __restrict__ in, float* __restrict__ out, int H, int W) {
  size_t i = (size_t)blockIdx.x * blockDim.x + threadIdx.x;
  size_t tot = (size_t)2 * H * W;
  if (i >= tot) return;
  int x = (int)(i % W);
  size_t r = i / W;
  int y = (int)(r % H);
  int b = (int)(r / H);
  const float* p = in + (size_t)b * H * W;
  float m = -INFINITY;
#pragma unroll
  for (int d = -4; d <= 4; ++d) {
    int yy = y + d;
    if (yy >= 0 && yy < H) m = fmaxf(m, p[(size_t)yy * W + x]);
  }
  out[i] = m;
}

__global__ void k_eq(const float* __restrict__ a, const float* __restrict__ b,
                     float* __restrict__ o, size_t n) {
  size_t i = (size_t)blockIdx.x * blockDim.x + threadIdx.x;
  if (i < n) o[i] = (a[i] == b[i]) ? 1.f : 0.f;
}

__global__ void k_suppss(const float* __restrict__ S, const float* __restrict__ P2,
                         float* __restrict__ SS, size_t n) {
  size_t i = (size_t)blockIdx.x * blockDim.x + threadIdx.x;
  if (i < n) SS[i] = (P2[i] > 0.f) ? 0.f : S[i];
}

__global__ void k_updmask(float* __restrict__ M, const float* __restrict__ P2,
                          const float* __restrict__ SS, const float* __restrict__ P3, size_t n) {
  size_t i = (size_t)blockIdx.x * blockDim.x + threadIdx.x;
  if (i >= n) return;
  bool supp = P2[i] > 0.f;
  bool nm = (SS[i] == P3[i]);
  M[i] = ((M[i] != 0.f) || (nm && !supp)) ? 1.f : 0.f;
}

__global__ void k_final(const float* __restrict__ S, const float* __restrict__ M,
                        float* __restrict__ F, int H, int W) {
  size_t i = (size_t)blockIdx.x * blockDim.x + threadIdx.x;
  size_t tot = (size_t)2 * H * W;
  if (i >= tot) return;
  int x = (int)(i % W);
  int y = (int)((i / W) % H);
  float v = (M[i] != 0.f) ? S[i] : 0.f;
  if (y < 4 || x < 4 || y >= H - 4 || x >= W - 4) v = -1.f;
  F[i] = v;
}

// ---------------------------------------------------------------------------
// top-K: histogram -> threshold scan -> deterministic wave-ballot selection
// ---------------------------------------------------------------------------
__global__ void k_zero_hist(int* h) {
  int i = blockIdx.x * blockDim.x + threadIdx.x;
  if (i < 2 * NBINS) h[i] = 0;
}

__global__ void k_hist(const float* __restrict__ F, int* __restrict__ hist, int npix) {
  size_t i = (size_t)blockIdx.x * blockDim.x + threadIdx.x;
  if (i >= (size_t)2 * npix) return;
  int b = (int)(i / npix);
  float v = F[i];
  int bin = (int)((v + 1.0f) * (NBINS / 2.002f));
  bin = iclamp(bin, 0, NBINS - 1);
  atomicAdd(&hist[b * NBINS + bin], 1);
}

__global__ void k_scan(const int* __restrict__ hist, float* __restrict__ thr, int K) {
  if (threadIdx.x != 0) return;
  int b = blockIdx.x;
  int cum = 0, t = 0;
  for (int i = NBINS - 1; i >= 0; --i) {
    cum += hist[b * NBINS + i];
    if (cum >= K) { t = i; break; }
  }
  thr[b] = -1.0f + t * (2.002f / NBINS);
}

__global__ void k_select(const float* __restrict__ F, const float* __restrict__ thr,
                         float* __restrict__ out_kp, float* __restrict__ out_sc,
                         int H, int W, int K) {
  int b = blockIdx.x;
  int lane = threadIdx.x;
  float th = thr[b];
  int npix = H * W;  // multiple of 32
  const float* p = F + (size_t)b * npix;
  int count = 0;
  for (int base = 0; base < npix && count < K; base += 32) {
    int idx = base + lane;
    float v = p[idx];
    bool pred = v >= th;
    unsigned long long m = __ballot((int)pred);
    int pos = count + __popcll(m & ((1ull << lane) - 1ull));
    if (pred && pos < K) {
      out_sc[(size_t)b * K + pos] = v;
      out_kp[((size_t)b * K + pos) * 2 + 0] = (float)(idx % W);
      out_kp[((size_t)b * K + pos) * 2 + 1] = (float)(idx / W);
    }
    count += (int)__popcll(m);
  }
  for (int pos = count + lane; pos < K; pos += 32) {
    out_sc[(size_t)b * K + pos] = -1.f;
    out_kp[((size_t)b * K + pos) * 2 + 0] = 0.f;
    out_kp[((size_t)b * K + pos) * 2 + 1] = 0.f;
  }
}

// ---------------------------------------------------------------------------
// descriptor map L2-normalize per pixel (256 ch, f16 in place)
// ---------------------------------------------------------------------------
__global__ void k_norm_desc(_Float16* __restrict__ D, int npix) {
  int i = blockIdx.x * blockDim.x + threadIdx.x;
  if (i >= npix) return;
  _Float16* p = D + (size_t)i * 256;
  float s = 0.f;
  for (int c = 0; c < 256; ++c) { float v = (float)p[c]; s += v * v; }
  float inv = 1.f / fmaxf(sqrtf(s), 1e-12f);
  for (int c = 0; c < 256; ++c) p[c] = (_Float16)((float)p[c] * inv);
}

// ---------------------------------------------------------------------------
// bilinear sample descriptors at keypoints, renormalize, write f32 output
// ---------------------------------------------------------------------------
__global__ void k_sample(const _Float16* __restrict__ D, const float* __restrict__ kp,
                         float* __restrict__ out, int K) {
  int i = blockIdx.x * blockDim.x + threadIdx.x;
  if (i >= 2 * K) return;
  int b = i / K;
  float kx = kp[(size_t)i * 2 + 0], ky = kp[(size_t)i * 2 + 1];
  const int w = 96, h = 96;
  float fx = (kx - 3.5f) * (95.0f / 763.5f);
  float fy = (ky - 3.5f) * (95.0f / 763.5f);
  float x0f = floorf(fx), y0f = floorf(fy);
  float wx = fx - x0f, wy = fy - y0f;
  int x0i = iclamp((int)x0f, 0, w - 1);
  int x1i = iclamp((int)x0f + 1, 0, w - 1);
  int y0i = iclamp((int)y0f, 0, h - 1);
  int y1i = iclamp((int)y0f + 1, 0, h - 1);
  const _Float16* p00 = D + (((size_t)b * h + y0i) * w + x0i) * 256;
  const _Float16* p10 = D + (((size_t)b * h + y0i) * w + x1i) * 256;
  const _Float16* p01 = D + (((size_t)b * h + y1i) * w + x0i) * 256;
  const _Float16* p11 = D + (((size_t)b * h + y1i) * w + x1i) * 256;
  float w00 = (1.f - wx) * (1.f - wy), w10 = wx * (1.f - wy);
  float w01 = (1.f - wx) * wy,         w11 = wx * wy;
  float ss = 0.f;
  for (int c = 0; c < 256; ++c) {
    float v = w00 * (float)p00[c] + w10 * (float)p10[c] +
              w01 * (float)p01[c] + w11 * (float)p11[c];
    ss += v * v;
  }
  float inv = 1.f / fmaxf(sqrtf(ss), 1e-12f);
  float* o = out + (size_t)i * 256;
  for (int c = 0; c < 256; ++c) {
    float v = w00 * (float)p00[c] + w10 * (float)p10[c] +
              w01 * (float)p01[c] + w11 * (float)p11[c];
    o[c] = v * inv;
  }
}

// ---------------------------------------------------------------------------
extern "C" void kernel_launch(void* const* d_in, const int* in_sizes, int n_in,
                              void* d_out, int out_size, void* d_ws, size_t ws_size,
                              hipStream_t stream) {
  const float* img = (const float*)d_in[0];
  const float* w1a = (const float*)d_in[1];  const float* b1a = (const float*)d_in[2];
  const float* w1b = (const float*)d_in[3];  const float* b1b = (const float*)d_in[4];
  const float* w2a = (const float*)d_in[5];  const float* b2a = (const float*)d_in[6];
  const float* w2b = (const float*)d_in[7];  const float* b2b = (const float*)d_in[8];
  const float* w3a = (const float*)d_in[9];  const float* b3a = (const float*)d_in[10];
  const float* w3b = (const float*)d_in[11]; const float* b3b = (const float*)d_in[12];
  const float* w4a = (const float*)d_in[13]; const float* b4a = (const float*)d_in[14];
  const float* w4b = (const float*)d_in[15]; const float* b4b = (const float*)d_in[16];
  const float* wpa = (const float*)d_in[17]; const float* bpa = (const float*)d_in[18];
  const float* wpb = (const float*)d_in[19]; const float* bpb = (const float*)d_in[20];
  const float* wda = (const float*)d_in[21]; const float* bda = (const float*)d_in[22];
  const float* wdb = (const float*)d_in[23]; const float* bdb = (const float*)d_in[24];

  float* out_kp = (float*)d_out;           // (2,1024,2)
  float* out_sc = out_kp + 2 * KTOP * 2;   // (2,1024)
  float* out_ds = out_sc + 2 * KTOP;       // (2,1024,256)

  char* ws = (char*)d_ws;
  size_t off = 0;
  auto alloc = [&](size_t bytes) -> void* {
    void* p = ws + off;
    off = (off + bytes + 255) & ~(size_t)255;
    return p;
  };

  const int H0 = 768, W0 = 768;
  const size_t NPIX = (size_t)2 * H0 * W0;          // full-res pixels (both batches)
  size_t nA = (size_t)2 * H0 * W0 * 64;             // halves (largest activation)
  _Float16* A   = (_Float16*)alloc(nA * 2);
  _Float16* B   = (_Float16*)alloc(nA * 2);
  _Float16* raw = (_Float16*)alloc((size_t)2 * 96 * 96 * 65 * 2);
  _Float16* Dm  = (_Float16*)alloc((size_t)2 * 96 * 96 * 256 * 2);
  float* S  = (float*)alloc(NPIX * 4);
  float* M  = (float*)alloc(NPIX * 4);
  float* T1 = (float*)alloc(NPIX * 4);
  float* T2 = (float*)alloc(NPIX * 4);
  float* T3 = (float*)alloc(NPIX * 4);
  float* Fb = (float*)alloc(NPIX * 4);
  int*   hist = (int*)alloc((size_t)2 * NBINS * 4);
  float* thr  = (float*)alloc(2 * sizeof(float));

  auto coTilesPad = [](int Cout) { return (((Cout + 15) / 16 + 3) / 4) * 4; };
  auto fragAlloc = [&](int Cout, int Cin, int ks) -> _Float16* {
    int coTP = coTilesPad(Cout);
    int nC = Cin / 32;
    return (_Float16*)alloc((size_t)ks * ks * nC * coTP * 32 * 16 * 2);
  };
  _Float16* f1b = fragAlloc(64, 64, 3);
  _Float16* f2a = fragAlloc(64, 64, 3);
  _Float16* f2b = fragAlloc(64, 64, 3);
  _Float16* f3a = fragAlloc(128, 64, 3);
  _Float16* f3b = fragAlloc(128, 128, 3);
  _Float16* f4a = fragAlloc(128, 128, 3);
  _Float16* f4b = fragAlloc(128, 128, 3);
  _Float16* fpa = fragAlloc(256, 128, 3);
  _Float16* fpb = fragAlloc(65, 256, 1);
  _Float16* fda = fragAlloc(256, 128, 3);
  _Float16* fdb = fragAlloc(256, 256, 1);

  auto prep = [&](const float* w, _Float16* frag, int Cout, int Cin, int ks) {
    int coTP = coTilesPad(Cout), nC = Cin / 32;
    dim3 g(ks * ks * nC, coTP, 1);
    k_prep_wfrag<<<g, dim3(32), 0, stream>>>(w, frag, Cout, Cin, ks, coTP);
  };
  auto conv = [&](const _Float16* in, const _Float16* frag, const float* bias,
                  _Float16* out, int H, int W, int Cin, int Cout, int ks, int pad, int relu) {
    int coTP = coTilesPad(Cout);
    dim3 g((W / 32) * H / 4, coTP / 4, 2);   // 4 waves/block, 32px x 64co per wave
    k_conv_wmma<<<g, dim3(128), 0, stream>>>(in, frag, bias, out, H, W, Cin, Cout, ks, pad,
                                             relu, coTP);
  };
  auto pool = [&](const _Float16* in, _Float16* out, int Ho, int Wo, int C) {
    size_t tot = (size_t)2 * Ho * Wo * C;
    k_pool<<<dim3((unsigned)((tot + 255) / 256)), 256, 0, stream>>>(in, out, Ho, Wo, C);
  };
  unsigned ewg = (unsigned)((NPIX + 255) / 256);

  // ---- backbone ----
  k_conv1<<<dim3((unsigned)((2 * (size_t)H0 * W0 + 255) / 256)), 256, 0, stream>>>(
      img, w1a, b1a, A, H0, W0);
  prep(w1b, f1b, 64, 64, 3);  conv(A, f1b, b1b, B, 768, 768, 64, 64, 3, 1, 1);
  pool(B, A, 384, 384, 64);
  prep(w2a, f2a, 64, 64, 3);  conv(A, f2a, b2a, B, 384, 384, 64, 64, 3, 1, 1);
  prep(w2b, f2b, 64, 64, 3);  conv(B, f2b, b2b, A, 384, 384, 64, 64, 3, 1, 1);
  pool(A, B, 192, 192, 64);
  prep(w3a, f3a, 128, 64, 3); conv(B, f3a, b3a, A, 192, 192, 64, 128, 3, 1, 1);
  prep(w3b, f3b, 128, 128, 3);conv(A, f3b, b3b, B, 192, 192, 128, 128, 3, 1, 1);
  pool(B, A, 96, 96, 128);
  prep(w4a, f4a, 128, 128, 3);conv(A, f4a, b4a, B, 96, 96, 128, 128, 3, 1, 1);
  prep(w4b, f4b, 128, 128, 3);conv(B, f4b, b4b, A, 96, 96, 128, 128, 3, 1, 1);  // feat -> A

  // ---- score head ----
  prep(wpa, fpa, 256, 128, 3); conv(A, fpa, bpa, B, 96, 96, 128, 256, 3, 1, 1);
  prep(wpb, fpb, 65, 256, 1);  conv(B, fpb, bpb, raw, 96, 96, 256, 65, 1, 0, 0);
  k_softmax_shuffle<<<dim3((2 * 96 * 96 + 255) / 256), 256, 0, stream>>>(raw, S, 96, 96);

  // ---- NMS (r=4) ----
  k_rowmax<<<ewg, 256, 0, stream>>>(S, T1, 768, 768);
  k_colmax<<<ewg, 256, 0, stream>>>(T1, T2, 768, 768);
  k_eq<<<ewg, 256, 0, stream>>>(S, T2, M, NPIX);
  for (int it = 0; it < 2; ++it) {
    k_rowmax<<<ewg, 256, 0, stream>>>(M, T1, 768, 768);
    k_colmax<<<ewg, 256, 0, stream>>>(T1, T2, 768, 768);   // T2 = mp(M)
    k_suppss<<<ewg, 256, 0, stream>>>(S, T2, Fb, NPIX);    // Fb = SS
    k_rowmax<<<ewg, 256, 0, stream>>>(Fb, T1, 768, 768);
    k_colmax<<<ewg, 256, 0, stream>>>(T1, T3, 768, 768);   // T3 = mp(SS)
    k_updmask<<<ewg, 256, 0, stream>>>(M, T2, Fb, T3, NPIX);
  }
  k_final<<<ewg, 256, 0, stream>>>(S, M, Fb, 768, 768);

  // ---- top-K ----
  k_zero_hist<<<dim3((2 * NBINS + 255) / 256), 256, 0, stream>>>(hist);
  k_hist<<<ewg, 256, 0, stream>>>(Fb, hist, H0 * W0);
  k_scan<<<dim3(2), 32, 0, stream>>>(hist, thr, KTOP);
  k_select<<<dim3(2), 32, 0, stream>>>(Fb, thr, out_kp, out_sc, 768, 768, KTOP);

  // ---- descriptor head ----
  prep(wda, fda, 256, 128, 3); conv(A, fda, bda, B, 96, 96, 128, 256, 3, 1, 1);
  prep(wdb, fdb, 256, 256, 1); conv(B, fdb, bdb, Dm, 96, 96, 256, 256, 1, 0, 0);
  k_norm_desc<<<dim3((2 * 96 * 96 + 255) / 256), 256, 0, stream>>>(Dm, 2 * 96 * 96);
  k_sample<<<dim3((2 * KTOP + 255) / 256), 256, 0, stream>>>(Dm, out_kp, out_ds, KTOP);

  (void)in_sizes; (void)n_in; (void)out_size; (void)ws_size;
}
